// GlobalAttention_81260781241005
// MI455X (gfx1250) — compile-verified, matrix operand order checked by reading, not run on hardware
//
#include <hip/hip_runtime.h>
#include <math.h>

// ---------------------------------------------------------------------------
// Types for CDNA5 WMMA (wave32): v_wmma_f32_16x16x32_bf16
// ---------------------------------------------------------------------------
typedef __attribute__((ext_vector_type(16))) __bf16 v16bf;
typedef __attribute__((ext_vector_type(8)))  float  v8f;

#define LDS_STRIDE 40   // 128 rows x 40 bf16 (80B pitch) keeps 16B frag loads bank-spread
#define TILE_ELEMS (128 * LDS_STRIDE)        // elements per 128x32 staged tile
#define TILE_BYTES (TILE_ELEMS * 2)

// Problem constants
#define BB   16
#define TT   512
#define SS   1024
#define DD   1024

// ---------------------------------------------------------------------------
// CDNA5 async global->LDS copy (ASYNCcnt-tracked), 16 bytes per lane.
// LDS offset operand is the wave-relative LDS byte address (low 32 bits of a
// generic pointer to __shared__, per the ISA flat->LDS mapping).
// ---------------------------------------------------------------------------
__device__ __forceinline__ void async_ld128(const void* gaddr, unsigned int ldsoff) {
  asm volatile("global_load_async_to_lds_b128 %0, %1, off"
               :
               : "v"(ldsoff), "v"((unsigned long long)(uintptr_t)gaddr)
               : "memory");
}
__device__ __forceinline__ void wait_async0() {
  asm volatile("s_wait_asynccnt 0x0" ::: "memory");
}

// ---------------------------------------------------------------------------
// LDS fragment load: 16x32 bf16 tile (row-major in LDS, pitch LDS_STRIDE).
// ISA 16-bit A layout: lanes 0-15 -> row = lane, elems 0..7 = K 0..7,
// elems 8..15 = K 16..23; lanes 16-31 -> same rows, K 8..15 / 24..31.
// B uses the symmetric layout with N in place of M.
// ---------------------------------------------------------------------------
__device__ __forceinline__ v16bf frag_ld(const __bf16* base, int lane) {
  int row  = lane & 15;
  int koff = (lane & 16) ? 8 : 0;
  const __bf16* p = base + row * LDS_STRIDE;
  union { v16bf v; uint4 q[2]; } u;
  u.q[0] = *reinterpret_cast<const uint4*>(p + koff);
  u.q[1] = *reinterpret_cast<const uint4*>(p + 16 + koff);
  return u.v;
}

// Synchronous 128x32 row-major staging (K contiguous in global).
__device__ __forceinline__ void load_tile_rowmajor(__bf16* smem, const __bf16* g,
                                                   size_t ld, int tid) {
#pragma unroll
  for (int i = 0; i < 2; ++i) {
    int idx  = tid + i * 256;
    int row  = idx >> 2;
    int quad = idx & 3;
    *reinterpret_cast<uint4*>(&smem[row * LDS_STRIDE + quad * 8]) =
        *reinterpret_cast<const uint4*>(&g[(size_t)row * ld + quad * 8]);
  }
}

// Async 128x32 row-major staging: 512 x 16B chunks across 256 threads.
__device__ __forceinline__ void stage_async_tile(unsigned int lds_base,
                                                 const __bf16* g, size_t ld, int tid) {
#pragma unroll
  for (int i = 0; i < 2; ++i) {
    int idx  = tid + i * 256;
    int row  = idx >> 2;
    int quad = idx & 3;
    async_ld128(g + (size_t)row * ld + quad * 8,
                lds_base + (unsigned int)((row * LDS_STRIDE + quad * 8) * 2));
  }
}

// Load a 32(k) x 128(n) tile (n contiguous in global) and transpose into
// smem[n][k]. g must point at (k0, n0).
__device__ __forceinline__ void load_tile_transpose(__bf16* smem, const __bf16* g,
                                                    size_t ld, int tid) {
  int k    = tid & 31;
  int nseg = tid >> 5;  // 0..7, 16 n each
  const __bf16* p = g + (size_t)k * ld + nseg * 16;
  union { uint4 q[2]; __bf16 e[16]; } u;
  u.q[0] = *reinterpret_cast<const uint4*>(p);
  u.q[1] = *reinterpret_cast<const uint4*>(p + 8);
#pragma unroll
  for (int j = 0; j < 16; ++j)
    smem[(nseg * 16 + j) * LDS_STRIDE + k] = u.e[j];
}

__device__ __forceinline__ void wave_coords(int& lane, int& wm, int& wn) {
  int tid = threadIdx.x;
  lane = tid & 31;
  int w = tid >> 5;
  wm = w >> 2;  // 0..1 : 64-row slab
  wn = w & 3;   // 0..3 : 32-col slab
}

__device__ __forceinline__ void mma_tile(const __bf16* Ac, const __bf16* Bc,
                                         int lane, int wm, int wn, v8f acc[4][2]) {
  v16bf af[4], bfr[2];
#pragma unroll
  for (int mi = 0; mi < 4; ++mi)
    af[mi] = frag_ld(Ac + (wm * 64 + mi * 16) * LDS_STRIDE, lane);
#pragma unroll
  for (int nj = 0; nj < 2; ++nj)
    bfr[nj] = frag_ld(Bc + (wn * 32 + nj * 16) * LDS_STRIDE, lane);
#pragma unroll
  for (int mi = 0; mi < 4; ++mi)
#pragma unroll
    for (int nj = 0; nj < 2; ++nj)
      acc[mi][nj] = __builtin_amdgcn_wmma_f32_16x16x32_bf16(
          false, af[mi], false, bfr[nj], (short)0, acc[mi][nj], false, false);
}

// ---------------------------------------------------------------------------
// Double-buffered NT GEMM core with async global->LDS staging.
// B is (N,K) row-major (K contiguous). A pre-offset to (m0,0), B to (n0,0).
// As/Bs must each hold 2*TILE_ELEMS.
// ---------------------------------------------------------------------------
__device__ __forceinline__ void gemm_core_nt_async(const __bf16* A, size_t lda,
                                                   const __bf16* Bm, size_t ldb,
                                                   int K, __bf16* As, __bf16* Bs,
                                                   v8f acc[4][2]) {
  int lane, wm, wn;
  wave_coords(lane, wm, wn);
  int tid = threadIdx.x;
  unsigned int asOff = (unsigned int)(uintptr_t)As;
  unsigned int bsOff = (unsigned int)(uintptr_t)Bs;

  stage_async_tile(asOff, A, lda, tid);
  stage_async_tile(bsOff, Bm, ldb, tid);

  int cur = 0;
  for (int k0 = 0; k0 < K; k0 += 32) {
    wait_async0();
    __syncthreads();
    int nxt = cur ^ 1;
    if (k0 + 32 < K) {
      stage_async_tile(asOff + (unsigned int)nxt * TILE_BYTES, A + k0 + 32, lda, tid);
      stage_async_tile(bsOff + (unsigned int)nxt * TILE_BYTES, Bm + k0 + 32, ldb, tid);
    }
    mma_tile(As + cur * TILE_ELEMS, Bs + cur * TILE_ELEMS, lane, wm, wn, acc);
    cur = nxt;
  }
}

// Synchronous NN GEMM core (B is (K,N) row-major, transposed while staging).
__device__ __forceinline__ void gemm_core_nn_sync(const __bf16* A, size_t lda,
                                                  const __bf16* Bm, size_t ldb,
                                                  int K, __bf16* As, __bf16* Bs,
                                                  v8f acc[4][2]) {
  int lane, wm, wn;
  wave_coords(lane, wm, wn);
  int tid = threadIdx.x;
  for (int k0 = 0; k0 < K; k0 += 32) {
    __syncthreads();
    load_tile_rowmajor(As, A + k0, lda, tid);
    load_tile_transpose(Bs, Bm + (size_t)k0 * ldb, ldb, tid);
    __syncthreads();
    mma_tile(As, Bs, lane, wm, wn, acc);
  }
}

__device__ __forceinline__ void acc_zero(v8f acc[4][2]) {
  v8f z = {0.f, 0.f, 0.f, 0.f, 0.f, 0.f, 0.f, 0.f};
#pragma unroll
  for (int mi = 0; mi < 4; ++mi)
#pragma unroll
    for (int nj = 0; nj < 2; ++nj) acc[mi][nj] = z;
}

// C/D fragment layout: VGPR r -> M = r (lanes 0-15) or r+8 (lanes 16-31),
// N = lane & 15. Calls f(mloc, nloc, val) for each element this lane owns.
template <typename F>
__device__ __forceinline__ void epilogue_foreach(const v8f acc[4][2], F f) {
  int tid = threadIdx.x, lane = tid & 31, w = tid >> 5;
  int wm = w >> 2, wn = w & 3;
  int mhi = (lane >> 4) << 3, nlo = lane & 15;
#pragma unroll
  for (int mi = 0; mi < 4; ++mi)
#pragma unroll
    for (int nj = 0; nj < 2; ++nj)
#pragma unroll
      for (int r = 0; r < 8; ++r) {
        int mloc = wm * 64 + mi * 16 + r + mhi;
        int nloc = wn * 32 + nj * 16 + nlo;
        f(mloc, nloc, acc[mi][nj][r]);
      }
}

// ---------------------------------------------------------------------------
// Kernels
// ---------------------------------------------------------------------------
__global__ void __launch_bounds__(256)
k_cvt_f32_bf16(const float* __restrict__ in, __bf16* __restrict__ out, int n) {
  int i = blockIdx.x * 256 + threadIdx.x;
  if (i < n) out[i] = (__bf16)in[i];
}

// GEMM 1: q_proj = Q (8192x1024) * W_in^T (N,K row-major) -> bf16
__global__ void __launch_bounds__(256)
k_gemm_qproj(const __bf16* __restrict__ Q, const __bf16* __restrict__ Win,
             __bf16* __restrict__ QP) {
  __shared__ __bf16 As[2 * TILE_ELEMS];
  __shared__ __bf16 Bs[2 * TILE_ELEMS];
  int m0 = blockIdx.y * 128, n0 = blockIdx.x * 128;
  v8f acc[4][2];
  acc_zero(acc);
  gemm_core_nt_async(Q + (size_t)m0 * DD, DD, Win + (size_t)n0 * DD, DD, DD,
                     As, Bs, acc);
  epilogue_foreach(acc, [&](int mloc, int nloc, float val) {
    QP[(size_t)(m0 + mloc) * DD + (n0 + nloc)] = (__bf16)val;
  });
}

// GEMM 2: align[b] = q_proj[b] (512x1024) * K[b]^T -> f32 raw scores written
// directly into the (t, b, s) output layout.
__global__ void __launch_bounds__(256)
k_gemm_align(const __bf16* __restrict__ QP, const __bf16* __restrict__ Kb,
             float* __restrict__ alignOut) {
  __shared__ __bf16 As[2 * TILE_ELEMS];
  __shared__ __bf16 Bs[2 * TILE_ELEMS];
  int b = blockIdx.z;
  int m0 = blockIdx.y * 128, n0 = blockIdx.x * 128;
  const __bf16* A = QP + (size_t)b * TT * DD + (size_t)m0 * DD;
  const __bf16* B = Kb + (size_t)b * SS * DD + (size_t)n0 * DD;
  v8f acc[4][2];
  acc_zero(acc);
  gemm_core_nt_async(A, DD, B, DD, DD, As, Bs, acc);
  epilogue_foreach(acc, [&](int mloc, int nloc, float val) {
    int t = m0 + mloc;
    int s = n0 + nloc;
    alignOut[((size_t)t * BB + b) * SS + s] = val;
  });
}

// Masked softmax over S=1024, one 256-thread block per (b, t) row.
// Reads/writes raw scores in place in d_out (t,b,s) and emits bf16 probs.
__global__ void __launch_bounds__(256)
k_softmax(float* __restrict__ alignOut, __bf16* __restrict__ Pbf,
          const long long* __restrict__ lens) {
  int rowId = blockIdx.x;         // b*512 + t
  int b = rowId >> 9, t = rowId & 511;
  long long len = lens[b];
  float* row = alignOut + ((size_t)t * BB + b) * SS;
  int tid = threadIdx.x;

  float v[4];
  float mx = -3.4e38f;
#pragma unroll
  for (int i = 0; i < 4; ++i) {
    int s = tid + i * 256;
    float x = row[s];
    x = (s < len) ? x : -INFINITY;
    v[i] = x;
    mx = fmaxf(mx, x);
  }
  __shared__ float red[256];
  red[tid] = mx;
  __syncthreads();
  for (int off = 128; off > 0; off >>= 1) {
    if (tid < off) red[tid] = fmaxf(red[tid], red[tid + off]);
    __syncthreads();
  }
  mx = red[0];
  __syncthreads();

  float e[4], sum = 0.f;
#pragma unroll
  for (int i = 0; i < 4; ++i) {
    e[i] = __expf(v[i] - mx);   // exp(-inf) == 0 handles the mask
    sum += e[i];
  }
  red[tid] = sum;
  __syncthreads();
  for (int off = 128; off > 0; off >>= 1) {
    if (tid < off) red[tid] += red[tid + off];
    __syncthreads();
  }
  float inv = 1.0f / red[0];
#pragma unroll
  for (int i = 0; i < 4; ++i) {
    int s = tid + i * 256;
    float p = e[i] * inv;
    row[s] = p;
    Pbf[(size_t)rowId * SS + s] = (__bf16)p;
  }
}

// GEMM 3: c[b] = P[b] (512x1024) * K[b] (1024x1024, N contiguous) -> bf16
__global__ void __launch_bounds__(256)
k_gemm_ctx(const __bf16* __restrict__ P, const __bf16* __restrict__ Kb,
           __bf16* __restrict__ C) {
  __shared__ __bf16 As[TILE_ELEMS];
  __shared__ __bf16 Bs[TILE_ELEMS];
  int b = blockIdx.z;
  int m0 = blockIdx.y * 128, n0 = blockIdx.x * 128;
  const __bf16* A = P + (size_t)b * TT * SS + (size_t)m0 * SS;
  const __bf16* B = Kb + (size_t)b * SS * DD + n0;   // column offset, NN form
  v8f acc[4][2];
  acc_zero(acc);
  gemm_core_nn_sync(A, SS, B, DD, SS, As, Bs, acc);
  epilogue_foreach(acc, [&](int mloc, int nloc, float val) {
    C[((size_t)b * TT + m0 + mloc) * DD + (n0 + nloc)] = (__bf16)val;
  });
}

// GEMM 4: attn_h = tanh(c * Wc^T + q * Wq^T), stored permuted (t, b, d).
// W_out is (D, 2D) row-major: Wc = cols 0..D-1, Wq = cols D..2D-1.
__global__ void __launch_bounds__(256)
k_gemm_out(const __bf16* __restrict__ C, const __bf16* __restrict__ Q,
           const __bf16* __restrict__ Wout, float* __restrict__ outH) {
  __shared__ __bf16 As[2 * TILE_ELEMS];
  __shared__ __bf16 Bs[2 * TILE_ELEMS];
  int m0 = blockIdx.y * 128, n0 = blockIdx.x * 128;
  v8f acc[4][2];
  acc_zero(acc);
  gemm_core_nt_async(C + (size_t)m0 * DD, DD, Wout + (size_t)n0 * 2 * DD, 2 * DD,
                     DD, As, Bs, acc);
  gemm_core_nt_async(Q + (size_t)m0 * DD, DD, Wout + (size_t)n0 * 2 * DD + DD,
                     2 * DD, DD, As, Bs, acc);
  epilogue_foreach(acc, [&](int mloc, int nloc, float val) {
    int m = m0 + mloc;
    int b = m >> 9;
    int t = m & 511;
    outH[((size_t)t * BB + b) * DD + (n0 + nloc)] = tanhf(val);
  });
}

// ---------------------------------------------------------------------------
// Host launcher
// ---------------------------------------------------------------------------
extern "C" void kernel_launch(void* const* d_in, const int* in_sizes, int n_in,
                              void* d_out, int out_size, void* d_ws, size_t ws_size,
                              hipStream_t stream) {
  (void)in_sizes; (void)n_in; (void)out_size; (void)ws_size;
  const float*     dQ    = (const float*)d_in[0];      // (16,512,1024)
  const float*     dMem  = (const float*)d_in[1];      // (16,1024,1024)
  const long long* dLen  = (const long long*)d_in[2];  // (16,) int64
  const float*     dWin  = (const float*)d_in[3];      // (1024,1024)
  const float*     dWout = (const float*)d_in[4];      // (1024,2048)

  float* outH     = (float*)d_out;                          // (512,16,1024)
  float* outAlign = outH + (size_t)TT * BB * DD;            // (512,16,1024)

  char* ws = (char*)d_ws;
  __bf16* Qbf    = (__bf16*)(ws);                         // 16 MB
  __bf16* Kbf    = (__bf16*)(ws + (size_t)(16 << 20));    // 32 MB
  __bf16* Winbf  = (__bf16*)(ws + (size_t)(48 << 20));    //  2 MB
  __bf16* Woutbf = (__bf16*)(ws + (size_t)(50 << 20));    //  4 MB
  __bf16* QPbf   = (__bf16*)(ws + (size_t)(54 << 20));    // 16 MB (reused as Cbf)
  __bf16* Pbf    = (__bf16*)(ws + (size_t)(70 << 20));    // 16 MB

  const int nQ = BB * TT * DD;      // 8388608
  const int nK = BB * SS * DD;      // 16777216
  const int nWi = DD * DD;          // 1048576
  const int nWo = DD * 2 * DD;      // 2097152

  k_cvt_f32_bf16<<<(nQ + 255) / 256, 256, 0, stream>>>(dQ, Qbf, nQ);
  k_cvt_f32_bf16<<<(nK + 255) / 256, 256, 0, stream>>>(dMem, Kbf, nK);
  k_cvt_f32_bf16<<<(nWi + 255) / 256, 256, 0, stream>>>(dWin, Winbf, nWi);
  k_cvt_f32_bf16<<<(nWo + 255) / 256, 256, 0, stream>>>(dWout, Woutbf, nWo);

  // 1) q_proj = Q * W_in^T
  k_gemm_qproj<<<dim3(DD / 128, (BB * TT) / 128), 256, 0, stream>>>(Qbf, Winbf, QPbf);
  // 2) raw scores straight into output align region, (t,b,s) layout
  k_gemm_align<<<dim3(SS / 128, TT / 128, BB), 256, 0, stream>>>(QPbf, Kbf, outAlign);
  // 3) masked softmax in place + bf16 probs
  k_softmax<<<BB * TT, 256, 0, stream>>>(outAlign, Pbf, dLen);
  // 4) c = P * K   (Cbf aliases QPbf; q_proj is dead after step 2)
  k_gemm_ctx<<<dim3(DD / 128, TT / 128, BB), 256, 0, stream>>>(Pbf, Kbf, QPbf);
  // 5) attn_h = tanh(c*Wc^T + q*Wq^T), permuted store
  k_gemm_out<<<dim3(DD / 128, (BB * TT) / 128), 256, 0, stream>>>(QPbf, Qbf, Woutbf, outH);
}
